// TELloss_69045894250747
// MI455X (gfx1250) — matrix operand order repaired and checked by reading, not compiled, not run on hardware
//
#include <hip/hip_runtime.h>
#include <hip/hip_bf16.h>
#include <math.h>

typedef __attribute__((ext_vector_type(16))) _Float16 v16h;
typedef __attribute__((ext_vector_type(8)))  float    v8f;

#define EPSF 1e-12f
#define NPTS 1024   // 32x32 graph nodes
#define NBC  24     // B*C = 8*3
#define KSUB 16     // subspace width (>=5, WMMA tile width)
#define NITER 30    // subspace iterations
#define KT   (NPTS / 32)   // 32 k-tiles of depth 32
#define MT   (NPTS / 16)   // 64 m-tiles
#define MPB  4             // m-tiles per wave in the WMMA GEMM

// ---------------------------------------------------------------------------
// 1) Bilinear 128x128 -> 32x32, then g = f / max(|f|, eps) (+-1 signs).
// ---------------------------------------------------------------------------
__global__ void resize_sign_kernel(const float* __restrict__ feats,
                                   float* __restrict__ g) {
  int idx = blockIdx.x * blockDim.x + threadIdx.x;       // 0 .. 24575
  if (idx >= NBC * NPTS) return;
  int pix = idx & (NPTS - 1);
  int bc  = idx >> 10;                                   // b*3 + c
  int i = pix >> 5, j = pix & 31;
  float ys = fmaxf((i + 0.5f) * 4.0f - 0.5f, 0.0f);
  float xs = fmaxf((j + 0.5f) * 4.0f - 0.5f, 0.0f);
  int y0 = (int)floorf(ys); int x0 = (int)floorf(xs);
  float wy = ys - (float)y0, wx = xs - (float)x0;
  y0 = min(max(y0, 0), 127); int y1 = min(y0 + 1, 127);
  x0 = min(max(x0, 0), 127); int x1 = min(x0 + 1, 127);
  const float* p = feats + (size_t)bc * (128 * 128);
  float top = p[y0 * 128 + x0] * (1.f - wx) + p[y0 * 128 + x1] * wx;
  float bot = p[y1 * 128 + x0] * (1.f - wx) + p[y1 * 128 + x1] * wx;
  float f = top * (1.f - wy) + bot * wy;
  g[idx] = f / fmaxf(fabsf(f), EPSF);
}

// ---------------------------------------------------------------------------
// 2) Per graph: wmax = (max|g|)^2, d_i = sum_j relu(g_i g_j)/wmax,
//    dm_i = (d<eps) ? 1 : rsqrt(d).
// ---------------------------------------------------------------------------
__global__ __launch_bounds__(1024)
void degree_kernel(const float* __restrict__ g, float* __restrict__ dm,
                   float* __restrict__ wmaxA, int bc) {
  __shared__ float sg[NPTS];
  __shared__ float buf[NPTS];
  int t = threadIdx.x;
  sg[t]  = g[bc * NPTS + t];
  buf[t] = fabsf(sg[t]);
  __syncthreads();
  for (int s = 512; s > 0; s >>= 1) {
    if (t < s) buf[t] = fmaxf(buf[t], buf[t + s]);
    __syncthreads();
  }
  float ma = buf[0];
  float wmax = ma * ma;
  if (wmax <= 0.f) wmax = 1.f;
  __syncthreads();
  float gi = sg[t];
  float d = 0.f;
  for (int j = 0; j < NPTS; ++j) d += fmaxf(gi * sg[j], 0.f);
  d /= wmax;
  dm[bc * NPTS + t] = (d < EPSF) ? 1.f : rsqrtf(d);
  if (t == 0) wmaxA[bc] = wmax;
}

// ---------------------------------------------------------------------------
// 3) Normalized adjacency A = relu(g g^T)/wmax * dm_i dm_j (L = I - A).
//    Written twice:
//      * Asw : f16 in the exact WMMA A-fragment order
//              [(mTile*KT + kTile)*32 + lane][16 halves] -> one 32B load/lane
//      * Af32: plain row-major f32 for the accurate Rayleigh-Ritz pass.
//    Thread index IS the fragment index -> fully coalesced Asw stores.
// ---------------------------------------------------------------------------
__global__ void build_A_kernel(const float* __restrict__ g,
                               const float* __restrict__ dm,
                               const float* __restrict__ wmaxA,
                               _Float16* __restrict__ Asw,
                               float* __restrict__ Af32, int bc) {
  int fidx = blockIdx.x * blockDim.x + threadIdx.x;      // 0 .. 1M-1
  int e     = fidx & 15;
  int lane  = (fidx >> 4) & 31;
  int kTile = (fidx >> 9) & (KT - 1);
  int mTile = fidx >> 14;
  int grp = lane >> 4, lr = lane & 15;
  int v = e >> 1, h = e & 1;
  int kk = 16 * grp + e;
  int i = mTile * 16 + lr;            // row
  int j = kTile * 32 + kk;            // col
  float w = fmaxf(g[bc * NPTS + i] * g[bc * NPTS + j], 0.f) / wmaxA[bc];
  float a = w * dm[bc * NPTS + i] * dm[bc * NPTS + j];
  Asw[fidx] = (_Float16)a;
  Af32[(size_t)i * NPTS + j] = a;
}

// ---------------------------------------------------------------------------
// 4) Deterministic subspace seed.
// ---------------------------------------------------------------------------
__global__ void init_X_kernel(float* __restrict__ Y) {
  int idx = blockIdx.x * blockDim.x + threadIdx.x;       // 0 .. 16383
  if (idx >= NPTS * KSUB) return;
  unsigned i = (unsigned)(idx >> 4), n = (unsigned)(idx & 15);
  unsigned hsh = i * 1664525u + n * 1013904223u + 12345u;
  hsh ^= hsh >> 16; hsh *= 2246822519u; hsh ^= hsh >> 13;
  Y[idx] = ((float)(hsh & 0xFFFFu) / 32768.0f) - 1.0f;
}

// ---------------------------------------------------------------------------
// 5) Modified Gram-Schmidt over 16 columns of length 1024 (one 32-wave block).
//    Emits f32 X (row-major) and f16 X in WMMA B-fragment order:
//    Bsw[(kTile*32 + lane)*16 + e] holds X[kTile*32 + 16*(lane>>4) + e, lane&15].
// ---------------------------------------------------------------------------
__device__ __forceinline__ float block_sum(float v, float* buf) {
  int t = threadIdx.x;
  buf[t] = v; __syncthreads();
  for (int s = 512; s > 0; s >>= 1) {
    if (t < s) buf[t] += buf[t + s];
    __syncthreads();
  }
  float r = buf[0];
  __syncthreads();
  return r;
}

__global__ __launch_bounds__(1024)
void orthonormalize_kernel(const float* __restrict__ Y,
                           float* __restrict__ Xf, _Float16* __restrict__ Bsw) {
  __shared__ float buf[NPTS];
  int t = threadIdx.x;                                   // row index (= k)
  float col[KSUB];
  #pragma unroll
  for (int j = 0; j < KSUB; ++j) col[j] = Y[t * KSUB + j];
  for (int j = 0; j < KSUB; ++j) {
    float n2 = block_sum(col[j] * col[j], buf);
    float inv = rsqrtf(fmaxf(n2, 1e-20f));
    col[j] *= inv;
    for (int k = j + 1; k < KSUB; ++k) {
      float d = block_sum(col[j] * col[k], buf);
      col[k] -= d * col[j];
    }
  }
  int kTile = t >> 5, grp = (t >> 4) & 1, e = t & 15;
  #pragma unroll
  for (int n = 0; n < KSUB; ++n) {
    Xf[t * KSUB + n] = col[n];
    Bsw[(size_t)((kTile * 32) + (grp * 16 + n)) * 16 + e] = (_Float16)col[n];
  }
}

// ---------------------------------------------------------------------------
// 6) Y = (I + A) X, the hot loop. One wave per MPB=4 row tiles: per k-step,
//    one 32B B-fragment load feeds 4 independent v_wmma_f32_16x16x32_f16.
//    All operands land as contiguous global_load_b128 pairs per lane.
// ---------------------------------------------------------------------------
__global__ __launch_bounds__(32)
void gemm_MX_wmma_kernel(const v16h* __restrict__ Asw,
                         const v16h* __restrict__ Bsw,
                         const float* __restrict__ Xf,
                         float* __restrict__ Y) {
  const int mb   = blockIdx.x;            // 0..15 -> m tiles 4mb..4mb+3
  const int lane = threadIdx.x;           // EXEC all ones (WMMA requirement)
  v8f c[MPB];
  #pragma unroll
  for (int tmi = 0; tmi < MPB; ++tmi) c[tmi] = (v8f){};

  for (int kt = 0; kt < KT; ++kt) {
    v16h b = Bsw[kt * 32 + lane];
    #pragma unroll
    for (int tmi = 0; tmi < MPB; ++tmi) {
      v16h a = Asw[(size_t)(((MPB * mb + tmi) * KT + kt) * 32) + lane];
      c[tmi] = __builtin_amdgcn_wmma_f32_16x16x32_f16(
          /*neg_a=*/false, a, /*neg_b=*/false, b,
          /*c_mod=*/(short)0, c[tmi], /*reuse_a=*/false, /*reuse_b=*/false);
    }
  }
  // C layout: VGPR r -> (M=r, N=lane) lanes 0-15 ; (M=r+8, N=lane-16) 16-31.
  const int grp = lane >> 4, lr = lane & 15;
  #pragma unroll
  for (int tmi = 0; tmi < MPB; ++tmi) {
    #pragma unroll
    for (int r = 0; r < 8; ++r) {
      int orow = (MPB * mb + tmi) * 16 + r + 8 * grp;
      Y[orow * KSUB + lr] = c[tmi][r] + Xf[orow * KSUB + lr];  // (I + A) X
    }
  }
}

// ---------------------------------------------------------------------------
// 7) Accurate Rayleigh-Ritz: Z = A X (exact f32 A), S = I + X^T Z,
//    16x16 Jacobi eigensolve; lambda(L) = 2 - theta(I+A), ascending.
// ---------------------------------------------------------------------------
__global__ void AX_f32_kernel(const float* __restrict__ Af32,
                              const float* __restrict__ Xf,
                              float* __restrict__ Z) {
  int idx = blockIdx.x * blockDim.x + threadIdx.x;       // 0 .. 16383
  if (idx >= NPTS * KSUB) return;
  int i = idx >> 4, n = idx & 15;
  float acc = 0.f;
  for (int j = 0; j < NPTS; ++j)
    acc += Af32[(size_t)i * NPTS + j] * Xf[j * KSUB + n];
  Z[idx] = acc;
}

__global__ __launch_bounds__(256)
void S_kernel(const float* __restrict__ Xf, const float* __restrict__ Z,
              float* __restrict__ S) {
  int idx = threadIdx.x;                                 // 256 = 16x16
  int j = idx >> 4, n = idx & 15;
  float acc = (j == n) ? 1.0f : 0.0f;                    // S = I + X^T A X
  for (int i = 0; i < NPTS; ++i) acc += Xf[i * KSUB + j] * Z[i * KSUB + n];
  S[idx] = acc;
}

__global__ void jacobi16_kernel(const float* __restrict__ S,
                                float* __restrict__ evalsL, int bc) {
  if (threadIdx.x != 0 || blockIdx.x != 0) return;
  float a[KSUB][KSUB];
  for (int i = 0; i < KSUB; ++i)
    for (int j = 0; j < KSUB; ++j) a[i][j] = S[i * KSUB + j];
  for (int sweep = 0; sweep < 30; ++sweep) {
    for (int p = 0; p < KSUB - 1; ++p) {
      for (int q = p + 1; q < KSUB; ++q) {
        float apq = a[p][q];
        if (fabsf(apq) < 1e-12f) continue;
        float tau = (a[q][q] - a[p][p]) / (2.0f * apq);
        float tt = (tau >= 0.f ? 1.f : -1.f) /
                   (fabsf(tau) + sqrtf(1.f + tau * tau));
        float cc = rsqrtf(1.f + tt * tt);
        float ss = tt * cc;
        for (int r = 0; r < KSUB; ++r) {
          float arp = a[r][p], arq = a[r][q];
          a[r][p] = cc * arp - ss * arq;
          a[r][q] = ss * arp + cc * arq;
        }
        for (int r = 0; r < KSUB; ++r) {
          float apr = a[p][r], aqr = a[q][r];
          a[p][r] = cc * apr - ss * aqr;
          a[q][r] = ss * apr + cc * aqr;
        }
      }
    }
  }
  float ev[KSUB];
  for (int i = 0; i < KSUB; ++i) ev[i] = a[i][i];
  for (int i = 1; i < KSUB; ++i) {               // sort theta descending
    float key = ev[i]; int j = i - 1;
    while (j >= 0 && ev[j] < key) { ev[j + 1] = ev[j]; --j; }
    ev[j + 1] = key;
  }
  for (int q = 0; q < KSUB; ++q) evalsL[bc * KSUB + q] = 2.0f - ev[q];
}

// ---------------------------------------------------------------------------
// 8) landa1/2/3 = mean_{b,k<5} |eval_c0 - eval_c1| etc. -> 3 output floats.
// ---------------------------------------------------------------------------
__global__ void final3_kernel(const float* __restrict__ evalsL,
                              float* __restrict__ out) {
  if (threadIdx.x != 0 || blockIdx.x != 0) return;
  float s1 = 0.f, s2 = 0.f, s3 = 0.f;
  for (int b = 0; b < 8; ++b) {
    for (int k = 0; k < 5; ++k) {
      float e0 = evalsL[(b * 3 + 0) * KSUB + k];
      float e1 = evalsL[(b * 3 + 1) * KSUB + k];
      float e2 = evalsL[(b * 3 + 2) * KSUB + k];
      s1 += fabsf(e0 - e1);
      s2 += fabsf(e0 - e2);
      s3 += fabsf(e1 - e2);
    }
  }
  out[0] = s1 / 40.0f;
  out[1] = s2 / 40.0f;
  out[2] = s3 / 40.0f;
}

// ---------------------------------------------------------------------------
extern "C" void kernel_launch(void* const* d_in, const int* in_sizes, int n_in,
                              void* d_out, int out_size, void* d_ws,
                              size_t ws_size, hipStream_t stream) {
  (void)in_sizes; (void)n_in; (void)out_size; (void)ws_size;
  const float* feats = (const float*)d_in[0];            // (8,3,128,128) f32
  float* out = (float*)d_out;                            // 3 floats

  // workspace layout (~6.7 MB total); all fragment bases 32B-aligned
  float* g      = (float*)d_ws;                          // 24576 f
  float* dm     = g + NBC * NPTS;                        // 24576 f
  float* wmaxA  = dm + NBC * NPTS;                       // 32 f (24 used)
  float* evalsL = wmaxA + 32;                            // 384 f
  float* S      = evalsL + NBC * KSUB;                   // 256 f
  float* Xf     = S + 256;                               // 16384 f
  float* Y      = Xf + NPTS * KSUB;                      // 16384 f
  float* Z      = Y + NPTS * KSUB;                       // 16384 f
  float* Af32   = Z + NPTS * KSUB;                       // 1M f (4 MB)
  _Float16* Bsw = (_Float16*)(Af32 + NPTS * NPTS);       // 16384 halves
  _Float16* Asw = Bsw + NPTS * KSUB;                     // 1M halves (2 MB)

  resize_sign_kernel<<<(NBC * NPTS + 255) / 256, 256, 0, stream>>>(feats, g);

  for (int bc = 0; bc < NBC; ++bc) {
    degree_kernel<<<1, 1024, 0, stream>>>(g, dm, wmaxA, bc);
    build_A_kernel<<<(NPTS * NPTS) / 256, 256, 0, stream>>>(g, dm, wmaxA,
                                                            Asw, Af32, bc);
    init_X_kernel<<<(NPTS * KSUB) / 256, 256, 0, stream>>>(Y);
    orthonormalize_kernel<<<1, 1024, 0, stream>>>(Y, Xf, Bsw);
    for (int it = 0; it < NITER; ++it) {
      gemm_MX_wmma_kernel<<<MT / MPB, 32, 0, stream>>>(
          (const v16h*)Asw, (const v16h*)Bsw, Xf, Y);
      orthonormalize_kernel<<<1, 1024, 0, stream>>>(Y, Xf, Bsw);
    }
    AX_f32_kernel<<<(NPTS * KSUB) / 256, 256, 0, stream>>>(Af32, Xf, Z);
    S_kernel<<<1, 256, 0, stream>>>(Xf, Z, S);
    jacobi16_kernel<<<1, 1, 0, stream>>>(S, evalsL, bc);
  }
  final3_kernel<<<1, 1, 0, stream>>>(evalsL, out);
}
